// L2Error_15539191677466
// MI455X (gfx1250) — compile-verified, matrix operand order checked by reading, not run on hardware
//
#include <hip/hip_runtime.h>

// Problem constants (B, Q, N, K) from the reference.
#define NB 8
#define NQ 128
#define NN 8192
#define NK 512

#define ROWS_PER_STAGE 64   // emb rows staged in LDS at a time (8 stages)
#define ROW_DW 68           // row pitch in dwords (128 bf16 = 64 dw + 4 pad); 68%64=4 -> conflict-free half-wave

typedef __attribute__((ext_vector_type(16))) __bf16  v16bf;
typedef __attribute__((ext_vector_type(8)))  float   v8f;

union Frag {                 // 8 dwords == 16 bf16; lets ds_load_b128 land directly in WMMA regs
    uint4  q[2];
    unsigned u[8];
    v16bf  b;
};

__device__ __forceinline__ float bfbits_to_f(unsigned short h) {
    union { unsigned u; float f; } t; t.u = ((unsigned)h) << 16; return t.f;
}

// Split f32 into bf16 hi (truncate) + bf16 lo (residual); hi in [15:0], lo in [31:16].
__device__ __forceinline__ unsigned pack_split(float x) {
    union { float f; unsigned u; } a; a.f = x;
    unsigned hu = a.u & 0xFFFF0000u;
    union { unsigned u; float f; } h; h.u = hu;
    float r = x - h.f;
    union { float f; unsigned u; } bb; bb.f = r;
    return (hu >> 16) | (bb.u & 0xFFFF0000u);
}

__global__ __launch_bounds__(256)
void l2min_wmma_kernel(const float* __restrict__ ze,
                       const float* __restrict__ emb,
                       float* __restrict__ out)
{
    __shared__ unsigned s_hi[ROWS_PER_STAGE * ROW_DW];  // bf16 hi plane (2 per dword, q-contiguous)
    __shared__ unsigned s_lo[ROWS_PER_STAGE * ROW_DW];  // bf16 lo plane
    __shared__ float    s_sq[ROWS_PER_STAGE];           // emb_sq for staged rows

    const int tid   = threadIdx.x;
    const int lane  = tid & 31;
    const int wave  = tid >> 5;
    const int half  = lane >> 4;     // 0: lanes 0-15, 1: lanes 16-31
    const int lan16 = lane & 15;

    const int b = blockIdx.y;
    const int n = blockIdx.x * 128 + wave * 16 + lan16;   // this lane's output column

    // ---- Load B fragments (ze slab) once; split into bf16 hi/lo; accumulate ze_sq partial.
    // B 32x16 operand: element e of lane L holds q = 32*s + (e<8 ? e : e+8) + 8*(L>=16),
    // n = n0 + (L%16). Pairs (2j,2j+1) are consecutive q -> build packed dwords directly.
    Frag Bhi[4], Blo[4];
    float zesq = 0.f;
    #pragma unroll
    for (int s = 0; s < 4; ++s) {
        #pragma unroll
        for (int j = 0; j < 8; ++j) {
            int e0 = 2 * j;
            int q0 = 32 * s + (e0 < 8 ? e0 : e0 + 8) + 8 * half;
            float x0 = ze[((size_t)(b * NQ + q0    )) * NN + n];
            float x1 = ze[((size_t)(b * NQ + q0 + 1)) * NN + n];
            zesq += x0 * x0 + x1 * x1;
            unsigned p0 = pack_split(x0), p1 = pack_split(x1);
            Bhi[s].u[j] = (p0 & 0xFFFFu) | (p1 << 16);
            Blo[s].u[j] = (p0 >> 16)     | (p1 & 0xFFFF0000u);
        }
    }

    float m = 3.4e38f;   // running min over codebook entries

    for (int stage = 0; stage < NK / ROWS_PER_STAGE; ++stage) {
        // ---- Cooperatively stage 64 emb rows into LDS planes (coalesced; 2 q per thread-iter).
        for (int idx = tid; idx < ROWS_PER_STAGE * (NQ / 2); idx += 256) {
            int kk = idx >> 6, q2 = idx & 63;           // q2 = q/2
            const float* ep = emb + (size_t)(stage * ROWS_PER_STAGE + kk) * NQ + 2 * q2;
            float x0 = ep[0], x1 = ep[1];
            unsigned p0 = pack_split(x0), p1 = pack_split(x1);
            s_hi[kk * ROW_DW + q2] = (p0 & 0xFFFFu) | (p1 << 16);
            s_lo[kk * ROW_DW + q2] = (p0 >> 16)     | (p1 & 0xFFFF0000u);
        }
        __syncthreads();
        // ---- emb_sq for the staged rows (hi+lo reconstructs ~f32).
        if (tid < ROWS_PER_STAGE) {
            float ssum = 0.f;
            const unsigned* rh = &s_hi[tid * ROW_DW];
            const unsigned* rl = &s_lo[tid * ROW_DW];
            for (int q2 = 0; q2 < NQ / 2; ++q2) {
                unsigned wh = rh[q2], wl = rl[q2];
                float v0 = bfbits_to_f((unsigned short)(wh & 0xFFFFu))
                         + bfbits_to_f((unsigned short)(wl & 0xFFFFu));
                float v1 = bfbits_to_f((unsigned short)(wh >> 16))
                         + bfbits_to_f((unsigned short)(wl >> 16));
                ssum += v0 * v0 + v1 * v1;
            }
            s_sq[tid] = ssum;
        }
        __syncthreads();

        // ---- 4 row-tiles of 16 codebook rows; split-bf16 WMMA accumulation over Q=128.
        #pragma unroll
        for (int kt = 0; kt < ROWS_PER_STAGE / 16; ++kt) {
            const int kk = kt * 16 + lan16;             // A row (M) this lane supplies
            const unsigned* rowh = &s_hi[kk * ROW_DW];
            const unsigned* rowl = &s_lo[kk * ROW_DW];
            v8f acc = {};
            #pragma unroll
            for (int s = 0; s < 4; ++s) {
                // A 16x32 operand: elements 0-7 at q = qb..qb+7, 8-15 at q = qb+16..qb+23,
                // qb = 32*s + 8*half. In dwords (2 bf16 each): qd..qd+3 and qd+8..qd+11.
                const int qd = 16 * s + 4 * half;
                Frag Ahi, Alo;
                Ahi.q[0] = *(const uint4*)(rowh + qd);
                Ahi.q[1] = *(const uint4*)(rowh + qd + 8);
                Alo.q[0] = *(const uint4*)(rowl + qd);
                Alo.q[1] = *(const uint4*)(rowl + qd + 8);
                // a*b ~= ah*bh + ah*bl + al*bh  (drop al*bl, ~2^-16 relative)
                acc = __builtin_amdgcn_wmma_f32_16x16x32_bf16(false, Ahi.b, false, Bhi[s].b,
                                                              (short)0, acc, false, false);
                acc = __builtin_amdgcn_wmma_f32_16x16x32_bf16(false, Ahi.b, false, Blo[s].b,
                                                              (short)0, acc, false, false);
                acc = __builtin_amdgcn_wmma_f32_16x16x32_bf16(false, Alo.b, false, Bhi[s].b,
                                                              (short)0, acc, false, false);
            }
            // ---- Epilogue: C/D layout -> VGPR i holds row M = i + 8*half, col N = lane%16.
            #pragma unroll
            for (int i = 0; i < 8; ++i) {
                int kst = kt * 16 + i + 8 * half;
                m = fminf(m, s_sq[kst] - 2.0f * acc[i]);
            }
        }
        __syncthreads();   // before re-staging LDS
    }

    // Combine the two lane halves: min over all k, sum ze_sq partials over all q.
    float mo = __shfl_xor(m, 16, 32);
    float zo = __shfl_xor(zesq, 16, 32);
    float r  = fminf(m, mo) + (zesq + zo);
    if (half == 0) out[(size_t)b * NN + n] = r;
}

extern "C" void kernel_launch(void* const* d_in, const int* in_sizes, int n_in,
                              void* d_out, int out_size, void* d_ws, size_t ws_size,
                              hipStream_t stream) {
    (void)in_sizes; (void)n_in; (void)out_size; (void)d_ws; (void)ws_size;
    const float* ze  = (const float*)d_in[0];   // (B, Q, N) f32
    const float* emb = (const float*)d_in[1];   // (K, Q)   f32
    float* out = (float*)d_out;                 // (B, N)   f32
    dim3 grid(NN / 128, NB);                    // (64, 8) blocks of 256 threads (8 wave32s)
    l2min_wmma_kernel<<<grid, 256, 0, stream>>>(ze, emb, out);
}
